// p_coo_23338852287246
// MI455X (gfx1250) — matrix operand order repaired and checked by reading, not compiled, not run on hardware
//
#include <hip/hip_runtime.h>
#include <hip/hip_bf16.h>
#include <math.h>

// ---------------- problem constants (from reference) ----------------
#define NN      16384
#define IN_DIM  64
#define ZDIM    128
#define KNN     16
#define OUTD    3

typedef __attribute__((ext_vector_type(16))) __bf16 v16bf;
typedef __attribute__((ext_vector_type(8)))  float  v8f;

union FragBF {
    v16bf v;
    uint4 q[2];
};

__device__ __forceinline__ unsigned short f32_to_bf16(float f) {
    unsigned int u = __float_as_uint(f);
    unsigned int r = u + 0x7FFFu + ((u >> 16) & 1u);   // round-to-nearest-even
    return (unsigned short)(r >> 16);
}

// ---------------- kNN: top-16 nearest neighbors on x[:, :3] ----------------
__global__ void knn_kernel(const float* __restrict__ x, int* __restrict__ knn) {
    __shared__ float tx[256], ty[256], tz[256];
    const int i = blockIdx.x * 256 + threadIdx.x;
    const float xi = x[(size_t)i * IN_DIM + 0];
    const float yi = x[(size_t)i * IN_DIM + 1];
    const float zi = x[(size_t)i * IN_DIM + 2];

    float bd[KNN]; int bi[KNN];
#pragma unroll
    for (int t = 0; t < KNN; ++t) { bd[t] = INFINITY; bi[t] = -1; }

    for (int base = 0; base < NN; base += 256) {
        const int j = base + threadIdx.x;
        tx[threadIdx.x] = x[(size_t)j * IN_DIM + 0];
        ty[threadIdx.x] = x[(size_t)j * IN_DIM + 1];
        tz[threadIdx.x] = x[(size_t)j * IN_DIM + 2];
        __syncthreads();
        for (int jj = 0; jj < 256; ++jj) {
            const int jg = base + jj;
            const float dx = xi - tx[jj];
            const float dy = yi - ty[jj];
            const float dz = zi - tz[jj];
            const float d = dx * dx + dy * dy + dz * dz;
            if (jg != i && d < bd[KNN - 1]) {
                bd[KNN - 1] = d; bi[KNN - 1] = jg;
#pragma unroll
                for (int t = KNN - 1; t > 0; --t) {
                    if (bd[t] < bd[t - 1]) {
                        float td = bd[t]; bd[t] = bd[t - 1]; bd[t - 1] = td;
                        int ti = bi[t];   bi[t] = bi[t - 1]; bi[t - 1] = ti;
                    }
                }
            }
        }
        __syncthreads();
    }
#pragma unroll
    for (int t = 0; t < KNN; ++t) knn[(size_t)i * KNN + t] = bi[t];
}

// ---------------- fp32 -> bf16 elementwise convert ----------------
__global__ void cvt_bf16_kernel(const float* __restrict__ in,
                                unsigned short* __restrict__ out, int count) {
    int idx = blockIdx.x * blockDim.x + threadIdx.x;
    if (idx < count) out[idx] = f32_to_bf16(in[idx]);
}

// ---- pack weights: W [din x dout] fp32 -> Wt [doutPad x din] bf16 (transposed, zero-padded) ----
__global__ void pack_wt_kernel(const float* __restrict__ W, unsigned short* __restrict__ Wt,
                               int din, int dout, int doutPad) {
    int idx = blockIdx.x * blockDim.x + threadIdx.x;
    int total = doutPad * din;
    if (idx >= total) return;
    int n = idx / din;
    int kk = idx - n * din;
    float val = (n < dout) ? W[(size_t)kk * dout + n] : 0.0f;
    Wt[idx] = f32_to_bf16(val);
}

// ---------------- WMMA GEMM: C[rows x nact] = A[rows x din](bf16) @ Wt^T + bias ----------------
// One wave -> one 16x16 C tile. blockDim.x = 32 * (doutPad/16); grid.x = rows/16.
// Fragment layouts per CDNA5 ISA 7.12.2 (16-bit A 16x32; B 32x16; f32 C 16x16).
__global__ void gemm_wmma_bf16_kernel(const unsigned short* __restrict__ A,
                                      const unsigned short* __restrict__ Bt,
                                      const float* __restrict__ bias,
                                      float* __restrict__ C,
                                      int lda, int din, int nact, int ldc) {
    const int lane = threadIdx.x & 31;
    const int wave = threadIdx.x >> 5;
    const int half = lane >> 4;          // 0: lanes 0-15, 1: lanes 16-31
    const int l16  = lane & 15;

    const int m0 = blockIdx.x * 16;
    const int n0 = wave * 16;
    const int row = m0 + l16;            // A row for this lane
    const int col = n0 + l16;            // B column for this lane

    const unsigned short* arow = A  + (size_t)row * lda;
    const unsigned short* brow = Bt + (size_t)col * din;

    v8f acc = {};
    for (int kb = 0; kb < din; kb += 32) {
        FragBF a, b;
        // A: half 0 -> K = kb..kb+7 and kb+16..kb+23 ; half 1 -> +8 on both
        a.q[0] = *(const uint4*)(arow + kb + half * 8);
        a.q[1] = *(const uint4*)(arow + kb + 16 + half * 8);
        // B: half 0 -> K = kb..kb+7 and kb+16..kb+23 ; half 1 -> +8 on both
        b.q[0] = *(const uint4*)(brow + kb + half * 8);
        b.q[1] = *(const uint4*)(brow + kb + 16 + half * 8);
        acc = __builtin_amdgcn_wmma_f32_16x16x32_bf16(
            false, a.v, false, b.v, (short)0, acc, false, false);
    }

    const int n = n0 + l16;
    if (n < nact) {
        const float bv = bias[n];
#pragma unroll
        for (int vv = 0; vv < 8; ++vv) {
            const int m = m0 + vv + half * 8;    // C VGPR v: M = v (+8 for lanes 16-31)
            C[(size_t)m * ldc + n] = acc[vv] + bv;
        }
    }
}

// ---------------- attention: 16 lanes per node, softmax over 16 neighbors ----------------
__global__ void attn_kernel(const float* __restrict__ q, const float* __restrict__ k,
                            const float* __restrict__ v, const float* __restrict__ s,
                            const int* __restrict__ knn,
                            const float* __restrict__ residual,
                            float* __restrict__ outF, unsigned short* __restrict__ outBf,
                            int C, float scale, int doTanh) {
    const int tid  = blockIdx.x * blockDim.x + threadIdx.x;
    const int node = tid >> 4;
    const int j    = tid & 15;           // neighbor slot 0..15
    const int lane = threadIdx.x & 31;
    const int gb   = lane & 16;          // base lane of this 16-lane group

    const int srcj = knn[(size_t)node * KNN + j];

    // alpha_j = <q[node], k[srcj]> * scale
    const float* qp = q + (size_t)node * C;
    const float* kp = k + (size_t)srcj * C;
    float alpha = 0.0f;
    for (int d = 0; d < C; ++d) alpha += qp[d] * kp[d];
    alpha *= scale;

    // softmax over the 16-lane group (xor offsets < 16 stay inside aligned groups)
    float amax = alpha;
    for (int off = 1; off < 16; off <<= 1)
        amax = fmaxf(amax, __shfl_xor(amax, off, 32));
    const float ex = __expf(alpha - amax);
    float denom = ex;
    for (int off = 1; off < 16; off <<= 1)
        denom += __shfl_xor(denom, off, 32);
    const float w = ex / denom;

    // pre-gather all 16 weights / source ids (all lanes active -> shuffles valid)
    float wAll[16]; int sAll[16];
#pragma unroll
    for (int jj = 0; jj < 16; ++jj) {
        wAll[jj] = __shfl(w,    gb + jj, 32);
        sAll[jj] = __shfl(srcj, gb + jj, 32);
    }

    for (int d = j; d < C; d += 16) {
        float acc = s[(size_t)node * C + d];
        if (residual) acc += residual[(size_t)node * C + d];
#pragma unroll
        for (int jj = 0; jj < 16; ++jj)
            acc += wAll[jj] * v[(size_t)sAll[jj] * C + d];
        if (doTanh) acc = tanhf(acc);
        outF[(size_t)node * C + d] = acc;
        if (outBf) outBf[(size_t)node * C + d] = f32_to_bf16(acc);
    }
}

// ---------------- host-side orchestration ----------------
extern "C" void kernel_launch(void* const* d_in, const int* in_sizes, int n_in,
                              void* d_out, int out_size, void* d_ws, size_t ws_size,
                              hipStream_t stream) {
    (void)in_sizes; (void)n_in; (void)out_size; (void)ws_size;

    const float* x = (const float*)d_in[1];   // d_in[0] = t (unused by reference)
    // params: for layer li in {1,2,3}, for nm in {q,k,v,s}: W, b
    const float* W[3][4]; const float* B[3][4];
    int p = 2;
    for (int li = 0; li < 3; ++li)
        for (int nm = 0; nm < 4; ++nm) { W[li][nm] = (const float*)d_in[p++]; B[li][nm] = (const float*)d_in[p++]; }

    // workspace layout
    char* ws = (char*)d_ws;
    size_t off = 0;
    auto alloc = [&](size_t bytes) { void* pr = ws + off; off = (off + bytes + 255) & ~(size_t)255; return pr; };
    int*            knnIdx = (int*)           alloc((size_t)NN * KNN * 4);
    unsigned short* xbf    = (unsigned short*)alloc((size_t)NN * IN_DIM * 2);
    unsigned short* hbf    = (unsigned short*)alloc((size_t)NN * ZDIM * 2);
    float*          qF     = (float*)         alloc((size_t)NN * ZDIM * 4);
    float*          kF     = (float*)         alloc((size_t)NN * ZDIM * 4);
    float*          vF     = (float*)         alloc((size_t)NN * ZDIM * 4);
    float*          sF     = (float*)         alloc((size_t)NN * ZDIM * 4);
    float*          h1F    = (float*)         alloc((size_t)NN * ZDIM * 4);
    float*          h2F    = (float*)         alloc((size_t)NN * ZDIM * 4);
    unsigned short* wt[4];
    for (int i = 0; i < 4; ++i) wt[i] = (unsigned short*)alloc((size_t)ZDIM * ZDIM * 2);
    float* proj[4] = { qF, kF, vF, sF };

    const float sc128 = 0.08838834764831844f;   // 1/sqrt(128)
    const float sc3   = 0.5773502691896258f;    // 1/sqrt(3)

    // 1) kNN graph
    knn_kernel<<<NN / 256, 256, 0, stream>>>(x, knnIdx);

    // 2) x -> bf16
    {
        int cnt = NN * IN_DIM;
        cvt_bf16_kernel<<<(cnt + 255) / 256, 256, 0, stream>>>(x, xbf, cnt);
    }

    // ---- layer 1: din=64, dout=128 ----
    for (int nm = 0; nm < 4; ++nm) {
        int total = ZDIM * IN_DIM;
        pack_wt_kernel<<<(total + 255) / 256, 256, 0, stream>>>(W[0][nm], wt[nm], IN_DIM, ZDIM, ZDIM);
    }
    for (int nm = 0; nm < 4; ++nm)
        gemm_wmma_bf16_kernel<<<NN / 16, 32 * (ZDIM / 16), 0, stream>>>(
            xbf, wt[nm], B[0][nm], proj[nm], IN_DIM, IN_DIM, ZDIM, ZDIM);
    attn_kernel<<<(NN * 16) / 256, 256, 0, stream>>>(
        qF, kF, vF, sF, knnIdx, (const float*)nullptr, h1F, hbf, ZDIM, sc128, 1);

    // ---- layer 2: din=128, dout=128, residual h1, tanh ----
    for (int nm = 0; nm < 4; ++nm) {
        int total = ZDIM * ZDIM;
        pack_wt_kernel<<<(total + 255) / 256, 256, 0, stream>>>(W[1][nm], wt[nm], ZDIM, ZDIM, ZDIM);
    }
    for (int nm = 0; nm < 4; ++nm)
        gemm_wmma_bf16_kernel<<<NN / 16, 32 * (ZDIM / 16), 0, stream>>>(
            hbf, wt[nm], B[1][nm], proj[nm], ZDIM, ZDIM, ZDIM, ZDIM);
    attn_kernel<<<(NN * 16) / 256, 256, 0, stream>>>(
        qF, kF, vF, sF, knnIdx, h1F, h2F, hbf, ZDIM, sc128, 1);
    // hbf now holds bf16(h2); h2F holds fp32 h2

    // ---- layer 3: din=128, dout=3 (padded to 16), no residual, no tanh ----
    for (int nm = 0; nm < 4; ++nm) {
        int total = 16 * ZDIM;
        pack_wt_kernel<<<(total + 255) / 256, 256, 0, stream>>>(W[2][nm], wt[nm], ZDIM, OUTD, 16);
    }
    for (int nm = 0; nm < 4; ++nm)
        gemm_wmma_bf16_kernel<<<NN / 16, 32 * (16 / 16), 0, stream>>>(
            hbf, wt[nm], B[2][nm], proj[nm], ZDIM, ZDIM, OUTD, OUTD);
    attn_kernel<<<(NN * 16) / 256, 256, 0, stream>>>(
        qF, kF, vF, sF, knnIdx, (const float*)nullptr, (float*)d_out,
        (unsigned short*)nullptr, OUTD, sc3, 0);
}
